// Decoder_84628035600714
// MI455X (gfx1250) — compile-verified, matrix operand order checked by reading, not run on hardware
//
#include <hip/hip_runtime.h>
#include <math.h>

// ---------------- problem constants (match reference) ----------------
#define BB      8
#define LTQ     512
#define LSQ     512
#define DMODEL  512
#define NH      8
#define DKH     64
#define DVH     64
#define DFF     2048
#define NLAYER  6
#define NT      (BB * LTQ)      // 4096 tokens
#define EPSLN   1e-5f

// ---------------- WMMA / vector types ----------------
typedef __attribute__((ext_vector_type(16))) _Float16 v16h;
typedef __attribute__((ext_vector_type(8)))  _Float16 v8h;
typedef __attribute__((ext_vector_type(8)))  float    v8f;
typedef __attribute__((ext_vector_type(4)))  float    f32x4;

union CFrag { v8f v; float e[8]; };

#define BM 128
#define BN 64
#define BK 32

// fragment-order LDS addressing (inverse of the documented 16-bit A/B layout):
//   lane = (row&15) + 16*((k>>3)&1)        (row = M for A, N for B)
//   slot = (k&7) + ((k>>4)<<3)
// A-staging: thread p = tid>>7 covers k in {8p..8p+7} U {8p+16..8p+23}, which is
// exactly lane (row&15)+16p's full 16-slot fragment -> one contiguous v16h store.
// B-staging: one aligned 8-k group -> one contiguous v8h store.

// =====================================================================
// Generic batched GEMM:  C = alpha * A(MxK) * op(B)(KxN) [+ bias] [relu]
// Batch z decomposed as i = z/split, j = z%split with independent strides.
// f32 operands converted to f16 while staging to LDS (fragment order,
// ping-pong double buffered -> one barrier per K-step);
// f32 accumulate via v_wmma_f32_16x16x32_f16.
// Requires M%128==0, N%64==0, K%32==0, 16B-aligned rows for A (and B if TRANSB).
// =====================================================================
template <int TRANSB>
__launch_bounds__(256)
__global__ void gemm_wmma_f16(
    const float* __restrict__ A, int lda, long long sA1, long long sA2,
    const float* __restrict__ Bm, int ldb, long long sB1, long long sB2,
    float* __restrict__ C, int ldc, long long sC1, long long sC2,
    const float* __restrict__ bias,
    int M, int N, int K, float alpha, int relu, int split)
{
    __shared__ __align__(32) _Float16 AsF[2][8][32][16];   // [buf][M-subtile][lane][slot]
    __shared__ __align__(32) _Float16 BsF[2][4][32][16];   // [buf][N-subtile][lane][slot]

    const int z = blockIdx.z;
    const int bi = z / split, bj = z % split;
    A  += (long long)bi * sA1 + (long long)bj * sA2;
    Bm += (long long)bi * sB1 + (long long)bj * sB2;
    C  += (long long)bi * sC1 + (long long)bj * sC2;

    const int tid    = threadIdx.x;
    const int lane   = tid & 31;
    const int wave   = tid >> 5;
    const int wm     = wave & 3;   // M pair: subtiles 2wm, 2wm+1 (rows wm*32..+31)
    const int wn     = wave >> 2;  // N pair: subtiles 2wn, 2wn+1 (cols wn*32..+31)
    const int m0     = blockIdx.y * BM;
    const int n0     = blockIdx.x * BN;
    const int laneLo = lane & 15;
    const int laneHi = lane >> 4;

    // A staging: 128 rows x 2 k-parities; thread -> (row, parity)
    const int ar = tid & 127;          // A row within tile
    const int ap = tid >> 7;           // parity: 0 -> k {0..7,16..23}, 1 -> {8..15,24..31}
    // B staging: 64 cols x 4 k-groups of 8
    const int bn  = tid & 63;          // B col (consecutive tid -> consecutive n)
    const int bkg = tid >> 6;          // B k-group (0..3)

    const long long aRowOff = (long long)(m0 + ar) * lda;

    float av[16], bv[8];

    auto loadA = [&](int k0) {
        const float* p0 = A + aRowOff + k0 + ap * 8;
        f32x4 a0 = *(const f32x4*)p0;
        f32x4 a1 = *(const f32x4*)(p0 + 4);
        f32x4 a2 = *(const f32x4*)(p0 + 16);
        f32x4 a3 = *(const f32x4*)(p0 + 20);
#pragma unroll
        for (int j = 0; j < 4; ++j) {
            av[j] = a0[j]; av[4 + j] = a1[j]; av[8 + j] = a2[j]; av[12 + j] = a3[j];
        }
    };
    auto loadB = [&](int k0) {
        if constexpr (TRANSB) {
            const float* p0 = Bm + (long long)(n0 + bn) * ldb + k0 + bkg * 8;
            f32x4 b0 = *(const f32x4*)p0;
            f32x4 b1 = *(const f32x4*)(p0 + 4);
#pragma unroll
            for (int j = 0; j < 4; ++j) { bv[j] = b0[j]; bv[4 + j] = b1[j]; }
        } else {
#pragma unroll
            for (int j = 0; j < 8; ++j)
                bv[j] = Bm[(long long)(k0 + bkg * 8 + j) * ldb + (n0 + bn)];
        }
    };

    loadA(0);
    loadB(0);

    CFrag c00, c01, c10, c11;
#pragma unroll
    for (int r = 0; r < 8; ++r) { c00.e[r] = 0.f; c01.e[r] = 0.f; c10.e[r] = 0.f; c11.e[r] = 0.f; }

    const int ntiles = K / BK;
    for (int kt = 0; kt < ntiles; ++kt) {
        const int buf = kt & 1;
        // ---- store staged registers to LDS (fragment order, ping-pong buffer) ----
        {
            v16h pa;
#pragma unroll
            for (int j = 0; j < 16; ++j) pa[j] = (_Float16)av[j];
            *(v16h*)&AsF[buf][ar >> 4][(ar & 15) + 16 * ap][0] = pa;
            v8h pb;
#pragma unroll
            for (int j = 0; j < 8; ++j) pb[j] = (_Float16)bv[j];
            *(v8h*)&BsF[buf][bn >> 4][(bn & 15) + 16 * (bkg & 1)][(bkg >> 1) * 8] = pb;
        }
        __syncthreads();   // single barrier per K-step (double buffered)

        // ---- issue global loads for next tile (overlap with DS + WMMA) ----
        if (kt + 1 < ntiles) {
            loadA((kt + 1) * BK);
            loadB((kt + 1) * BK);
        }

        // ---- fragment loads: contiguous 32B per lane -> ds_load_b128 x2 each ----
        v16h a0 = *(const v16h*)&AsF[buf][2 * wm][lane][0];
        v16h a1 = *(const v16h*)&AsF[buf][2 * wm + 1][lane][0];
        v16h b0 = *(const v16h*)&BsF[buf][2 * wn][lane][0];
        v16h b1 = *(const v16h*)&BsF[buf][2 * wn + 1][lane][0];

        c00.v = __builtin_amdgcn_wmma_f32_16x16x32_f16(false, a0, false, b0, (short)0, c00.v, false, false);
        c01.v = __builtin_amdgcn_wmma_f32_16x16x32_f16(false, a0, false, b1, (short)0, c01.v, false, false);
        c10.v = __builtin_amdgcn_wmma_f32_16x16x32_f16(false, a1, false, b0, (short)0, c10.v, false, false);
        c11.v = __builtin_amdgcn_wmma_f32_16x16x32_f16(false, a1, false, b1, (short)0, c11.v, false, false);
    }

    // C layout: VGPR r, lanes 0-15 -> row r, lanes 16-31 -> row r+8; col = lane&15
    const int colA = n0 + 2 * wn * 16 + laneLo;
    const int colB = colA + 16;
    const int row0 = m0 + 2 * wm * 16 + laneHi * 8;
    const int row1 = row0 + 16;
    float bsA = 0.f, bsB = 0.f;
    if (bias) { bsA = bias[colA]; bsB = bias[colB]; }
#pragma unroll
    for (int r = 0; r < 8; ++r) {
        float v00 = alpha * c00.e[r] + bsA;
        float v01 = alpha * c01.e[r] + bsB;
        float v10 = alpha * c10.e[r] + bsA;
        float v11 = alpha * c11.e[r] + bsB;
        if (relu) {
            v00 = fmaxf(v00, 0.f); v01 = fmaxf(v01, 0.f);
            v10 = fmaxf(v10, 0.f); v11 = fmaxf(v11, 0.f);
        }
        C[(long long)(row0 + r) * ldc + colA] = v00;
        C[(long long)(row0 + r) * ldc + colB] = v01;
        C[(long long)(row1 + r) * ldc + colA] = v10;
        C[(long long)(row1 + r) * ldc + colB] = v11;
    }
}

// =====================================================================
// x[b,t,:] = tgt_emb[tgt_seq[b,t]] + pos_emb[tgt_pos[b,t]]
// =====================================================================
__global__ void embed_kernel(const int* __restrict__ seq, const int* __restrict__ pos,
                             const float* __restrict__ wemb, const float* __restrict__ pemb,
                             float* __restrict__ x)
{
    const int row = blockIdx.x;
    const int s = seq[row], p = pos[row];
    const int d = threadIdx.x;
    const float* we = wemb + (size_t)s * DMODEL;
    const float* pe = pemb + (size_t)p * DMODEL;
    float* xr = x + (size_t)row * DMODEL;
    xr[d]       = we[d]       + pe[d];
    xr[d + 256] = we[d + 256] + pe[d + 256];
}

// =====================================================================
// In-place masked softmax over rows of attn [NH*BB*LT, LS].
// causal!=0 adds the upper-triangular mask; seq gives PAD positions.
// =====================================================================
__global__ void softmax_mask_kernel(float* __restrict__ attn,
                                    const int* __restrict__ seq, int causal)
{
    const int row = blockIdx.x;                 // h*BB*LT + b*LT + q
    const int q = row % LTQ;
    const int b = (row / LTQ) % BB;
    float* p = attn + (size_t)row * LSQ;
    const int tid = threadIdx.x;
    const int s0 = tid, s1 = tid + 256;

    const bool m0 = (causal && s0 > q) || (seq[b * LSQ + s0] == 0);
    const bool m1 = (causal && s1 > q) || (seq[b * LSQ + s1] == 0);
    float v0 = m0 ? -INFINITY : p[s0];
    float v1 = m1 ? -INFINITY : p[s1];

    __shared__ float red[256];
    red[tid] = fmaxf(v0, v1);
    __syncthreads();
    for (int t = 128; t > 0; t >>= 1) {
        if (tid < t) red[tid] = fmaxf(red[tid], red[tid + t]);
        __syncthreads();
    }
    const float mx = red[0];
    __syncthreads();

    const float e0 = expf(v0 - mx), e1 = expf(v1 - mx);  // exp(-inf)=0
    red[tid] = e0 + e1;
    __syncthreads();
    for (int t = 128; t > 0; t >>= 1) {
        if (tid < t) red[tid] += red[tid + t];
        __syncthreads();
    }
    const float inv = 1.0f / red[0];
    p[s0] = e0 * inv;
    p[s1] = e1 * inv;
}

// =====================================================================
// x[row] = LayerNorm(x[row] + t[row]) * g + b   (in place; D=512)
// =====================================================================
__global__ void add_ln_kernel(float* __restrict__ x, const float* __restrict__ t,
                              const float* __restrict__ g, const float* __restrict__ bta)
{
    const int row = blockIdx.x;
    const int tid = threadIdx.x;
    float* xr = x + (size_t)row * DMODEL;
    const float* tr = t + (size_t)row * DMODEL;
    float a0 = xr[tid] + tr[tid];
    float a1 = xr[tid + 256] + tr[tid + 256];

    __shared__ float red[256];
    red[tid] = a0 + a1;
    __syncthreads();
    for (int s = 128; s > 0; s >>= 1) { if (tid < s) red[tid] += red[tid + s]; __syncthreads(); }
    const float mu = red[0] * (1.0f / DMODEL);
    __syncthreads();

    const float d0 = a0 - mu, d1 = a1 - mu;
    red[tid] = d0 * d0 + d1 * d1;
    __syncthreads();
    for (int s = 128; s > 0; s >>= 1) { if (tid < s) red[tid] += red[tid + s]; __syncthreads(); }
    const float rs = rsqrtf(red[0] * (1.0f / DMODEL) + EPSLN);

    xr[tid]       = d0 * rs * g[tid]       + bta[tid];
    xr[tid + 256] = d1 * rs * g[tid + 256] + bta[tid + 256];
}

// =====================================================================
static inline void launch_gemm(hipStream_t st,
    const float* A, int lda, long long sA1, long long sA2,
    const float* Bm, int ldb, long long sB1, long long sB2, int tB,
    float* C, int ldc, long long sC1, long long sC2,
    const float* bias, int M, int N, int K, float alpha, int relu,
    int batch, int split)
{
    dim3 g(N / BN, M / BM, batch), blk(256);
    if (tB)
        gemm_wmma_f16<1><<<g, blk, 0, st>>>(A, lda, sA1, sA2, Bm, ldb, sB1, sB2,
                                            C, ldc, sC1, sC2, bias, M, N, K, alpha, relu, split);
    else
        gemm_wmma_f16<0><<<g, blk, 0, st>>>(A, lda, sA1, sA2, Bm, ldb, sB1, sB2,
                                            C, ldc, sC1, sC2, bias, M, N, K, alpha, relu, split);
}

extern "C" void kernel_launch(void* const* d_in, const int* in_sizes, int n_in,
                              void* d_out, int out_size, void* d_ws, size_t ws_size,
                              hipStream_t stream)
{
    (void)in_sizes; (void)n_in; (void)out_size; (void)ws_size;

    const int*   tgt_seq = (const int*)d_in[0];
    const int*   tgt_pos = (const int*)d_in[1];
    const int*   src_seq = (const int*)d_in[2];
    const float* enc_out = (const float*)d_in[3];
    const float* wemb    = (const float*)d_in[4];
    const float* pemb    = (const float*)d_in[5];
    const float* slf_wq  = (const float*)d_in[6];
    const float* slf_wk  = (const float*)d_in[7];
    const float* slf_wv  = (const float*)d_in[8];
    const float* slf_pw  = (const float*)d_in[9];
    const float* slf_pb  = (const float*)d_in[10];
    const float* slf_g   = (const float*)d_in[11];
    const float* slf_b   = (const float*)d_in[12];
    const float* enc_wq  = (const float*)d_in[13];
    const float* enc_wk  = (const float*)d_in[14];
    const float* enc_wv  = (const float*)d_in[15];
    const float* enc_pw  = (const float*)d_in[16];
    const float* enc_pb  = (const float*)d_in[17];
    const float* enc_g   = (const float*)d_in[18];
    const float* enc_b   = (const float*)d_in[19];
    const float* ffn_w1  = (const float*)d_in[20];
    const float* ffn_b1  = (const float*)d_in[21];
    const float* ffn_w2  = (const float*)d_in[22];
    const float* ffn_b2  = (const float*)d_in[23];
    const float* ffn_g   = (const float*)d_in[24];
    const float* ffn_b   = (const float*)d_in[25];

    float* out = (float*)d_out;
    float* ws  = (float*)d_ws;

    const size_t SZ  = (size_t)NT * DMODEL;                 // 2,097,152 floats
    float* x   = ws;                                        // [NT, D]
    float* qb  = ws + 1 * SZ;                               // [NT, H*DK]
    float* kb  = ws + 2 * SZ;                               // [NT, H*DK]
    float* vb  = ws + 3 * SZ;                               // [NT, H*DV]
    float* ctx = ws + 4 * SZ;                               // [NT, H*DV]
    float* tmp = ws + 5 * SZ;                               // [NT, D]
    float* hid = ws + 6 * SZ;                               // [NT, DFF] = 4*SZ

    const size_t ATT = (size_t)NH * BB * LTQ * LSQ;         // 16,777,216 per layer
    float* slfA = out + SZ;                                 // [L, H*B, LT, LS]
    float* encA = out + SZ + (size_t)NLAYER * ATT;

    embed_kernel<<<dim3(NT), dim3(256), 0, stream>>>(tgt_seq, tgt_pos, wemb, pemb, x);

    const float iscale = 1.0f / sqrtf((float)DMODEL);
    const long long sHB_tok = (long long)LTQ * NH * DKH;    // per-batch token-block stride
    const long long sAtt_h  = (long long)BB * LTQ * LSQ;
    const long long sAtt_b  = (long long)LTQ * LSQ;

    for (int l = 0; l < NLAYER; ++l) {
        // ---------------- self attention ----------------
        const float* wq = slf_wq + (size_t)l * NH * DMODEL * DKH;
        const float* wk = slf_wk + (size_t)l * NH * DMODEL * DKH;
        const float* wv = slf_wv + (size_t)l * NH * DMODEL * DVH;
        float* attn = slfA + (size_t)l * ATT;

        launch_gemm(stream, x, DMODEL, 0, 0,
                    wq, DKH, (long long)DMODEL * DKH, 0, 0,
                    qb, NH * DKH, DKH, 0,
                    nullptr, NT, DKH, DMODEL, 1.0f, 0, NH, 1);
        launch_gemm(stream, x, DMODEL, 0, 0,
                    wk, DKH, (long long)DMODEL * DKH, 0, 0,
                    kb, NH * DKH, DKH, 0,
                    nullptr, NT, DKH, DMODEL, 1.0f, 0, NH, 1);
        launch_gemm(stream, x, DMODEL, 0, 0,
                    wv, DVH, (long long)DMODEL * DVH, 0, 0,
                    vb, NH * DVH, DVH, 0,
                    nullptr, NT, DVH, DMODEL, 1.0f, 0, NH, 1);

        launch_gemm(stream, qb, NH * DKH, DKH, sHB_tok,
                    kb, NH * DKH, DKH, sHB_tok, 1,
                    attn, LSQ, sAtt_h, sAtt_b,
                    nullptr, LTQ, LSQ, DKH, iscale, 0, NH * BB, BB);
        softmax_mask_kernel<<<dim3(NH * BB * LTQ), dim3(256), 0, stream>>>(attn, tgt_seq, 1);

        launch_gemm(stream, attn, LSQ, sAtt_h, sAtt_b,
                    vb, NH * DVH, DVH, sHB_tok, 0,
                    ctx, NH * DVH, DVH, sHB_tok,
                    nullptr, LTQ, DVH, LSQ, 1.0f, 0, NH * BB, BB);

        launch_gemm(stream, ctx, NH * DVH, 0, 0,
                    slf_pw + (size_t)l * NH * DVH * DMODEL, DMODEL, 0, 0, 0,
                    tmp, DMODEL, 0, 0,
                    slf_pb + (size_t)l * DMODEL, NT, DMODEL, NH * DVH, 1.0f, 0, 1, 1);
        add_ln_kernel<<<dim3(NT), dim3(256), 0, stream>>>(
            x, tmp, slf_g + (size_t)l * DMODEL, slf_b + (size_t)l * DMODEL);

        // ---------------- cross attention ----------------
        const float* cwq = enc_wq + (size_t)l * NH * DMODEL * DKH;
        const float* cwk = enc_wk + (size_t)l * NH * DMODEL * DKH;
        const float* cwv = enc_wv + (size_t)l * NH * DMODEL * DVH;
        float* attn2 = encA + (size_t)l * ATT;

        launch_gemm(stream, x, DMODEL, 0, 0,
                    cwq, DKH, (long long)DMODEL * DKH, 0, 0,
                    qb, NH * DKH, DKH, 0,
                    nullptr, NT, DKH, DMODEL, 1.0f, 0, NH, 1);
        launch_gemm(stream, enc_out, DMODEL, 0, 0,
                    cwk, DKH, (long long)DMODEL * DKH, 0, 0,
                    kb, NH * DKH, DKH, 0,
                    nullptr, NT, DKH, DMODEL, 1.0f, 0, NH, 1);
        launch_gemm(stream, enc_out, DMODEL, 0, 0,
                    cwv, DVH, (long long)DMODEL * DVH, 0, 0,
                    vb, NH * DVH, DVH, 0,
                    nullptr, NT, DVH, DMODEL, 1.0f, 0, NH, 1);

        launch_gemm(stream, qb, NH * DKH, DKH, sHB_tok,
                    kb, NH * DKH, DKH, sHB_tok, 1,
                    attn2, LSQ, sAtt_h, sAtt_b,
                    nullptr, LTQ, LSQ, DKH, iscale, 0, NH * BB, BB);
        softmax_mask_kernel<<<dim3(NH * BB * LTQ), dim3(256), 0, stream>>>(attn2, src_seq, 0);

        launch_gemm(stream, attn2, LSQ, sAtt_h, sAtt_b,
                    vb, NH * DVH, DVH, sHB_tok, 0,
                    ctx, NH * DVH, DVH, sHB_tok,
                    nullptr, LTQ, DVH, LSQ, 1.0f, 0, NH * BB, BB);

        launch_gemm(stream, ctx, NH * DVH, 0, 0,
                    enc_pw + (size_t)l * NH * DVH * DMODEL, DMODEL, 0, 0, 0,
                    tmp, DMODEL, 0, 0,
                    enc_pb + (size_t)l * DMODEL, NT, DMODEL, NH * DVH, 1.0f, 0, 1, 1);
        add_ln_kernel<<<dim3(NT), dim3(256), 0, stream>>>(
            x, tmp, enc_g + (size_t)l * DMODEL, enc_b + (size_t)l * DMODEL);

        // ---------------- FFN ----------------
        launch_gemm(stream, x, DMODEL, 0, 0,
                    ffn_w1 + (size_t)l * DMODEL * DFF, DFF, 0, 0, 0,
                    hid, DFF, 0, 0,
                    ffn_b1 + (size_t)l * DFF, NT, DFF, DMODEL, 1.0f, 1, 1, 1);
        launch_gemm(stream, hid, DFF, 0, 0,
                    ffn_w2 + (size_t)l * DFF * DMODEL, DMODEL, 0, 0, 0,
                    tmp, DMODEL, 0, 0,
                    ffn_b2 + (size_t)l * DMODEL, NT, DMODEL, DFF, 1.0f, 0, 1, 1);
        add_ln_kernel<<<dim3(NT), dim3(256), 0, stream>>>(
            x, tmp, ffn_g + (size_t)l * DMODEL, ffn_b + (size_t)l * DMODEL);
    }

    // final hidden state -> front of d_out
    hipMemcpyAsync(out, x, SZ * sizeof(float), hipMemcpyDeviceToDevice, stream);
}
